// SkipGram_9431748182542
// MI455X (gfx1250) — matrix operand order repaired and checked by reading, not compiled, not run on hardware
//
#include <hip/hip_runtime.h>

// SkipGram negative-sampling loss for MI455X (gfx1250, wave32).
// Scores for a 16-row batch tile = diagonal of C(16x128) x I(128x16),
// computed with v_wmma_f32_16x16x32_f16 (4 K-chunks of 32).
// Memory-bound gather workload (~0.5 FLOP/byte): both embedding tables
// (51.2 MB each) fit in the 192 MB L2, so gathers are L2-served after cold
// misses; WMMA makes the 21x16 dot products per tile essentially free.

typedef __attribute__((ext_vector_type(16))) _Float16 v16h;
typedef __attribute__((ext_vector_type(8)))  float    v8f;

#define EMB  128
#define NEGS 20
#define WAVES_PER_BLOCK 8
#define ROWS_PER_WAVE   16
#define ROWS_PER_BLOCK  (WAVES_PER_BLOCK * ROWS_PER_WAVE)   // 128

// Load 8 consecutive f32 (two global_load_b128) and convert into f16 slots
// [AT, AT+8) of a v16h WMMA operand fragment (lowers to v_cvt_pk_f16_f32).
template <int AT>
__device__ __forceinline__ void load8_cvt(v16h& dst, const float* __restrict__ p) {
  const float4* q = reinterpret_cast<const float4*>(p);   // 32B-aligned (8-float offsets)
  float4 f0 = q[0];
  float4 f1 = q[1];
  dst[AT + 0] = (_Float16)f0.x; dst[AT + 1] = (_Float16)f0.y;
  dst[AT + 2] = (_Float16)f0.z; dst[AT + 3] = (_Float16)f0.w;
  dst[AT + 4] = (_Float16)f1.x; dst[AT + 5] = (_Float16)f1.y;
  dst[AT + 6] = (_Float16)f1.z; dst[AT + 7] = (_Float16)f1.w;
}

__global__ void __launch_bounds__(256)
skipgram_nsloss_wmma(const int*   __restrict__ in_idx,
                     const int*   __restrict__ out_idx,
                     const int*   __restrict__ neg_idx,
                     const float* __restrict__ in_emb,
                     const float* __restrict__ out_emb,
                     float*       __restrict__ partial) {
  const int tid  = threadIdx.x;
  const int lane = tid & 31;
  const int wave = tid >> 5;
  const int kh   = lane >> 4;          // which K-half this lane supplies
  const int base = (blockIdx.x * WAVES_PER_BLOCK + wave) * ROWS_PER_WAVE;
  const int t    = base + (lane & 15); // batch row == tile row == tile column

  // ---- B operand: 16 input embeddings as columns (32x16 f16 per K-chunk).
  // lane n<16: column n, K = chunk*32 + 0..15 ; lane 16+n: K = chunk*32 + 16..31
  const int    ib   = in_idx[t];
  const float* brow = in_emb + (size_t)ib * EMB + kh * 16;
  v16h b[4];
  #pragma unroll
  for (int c = 0; c < 4; ++c) {
    load8_cvt<0>(b[c], brow + c * 32 + 0);
    load8_cvt<8>(b[c], brow + c * 32 + 8);
  }

  const int   ob        = out_idx[t];
  const float diag_mask = ((lane < 8) || (lane >= 24)) ? 1.0f : 0.0f;
  float acc = 0.0f;

  // ctx 0 = positive sample, ctx 1..20 = negatives.
  for (int ctx = 0; ctx <= NEGS; ++ctx) {
    int ridx = ob;
    if (ctx > 0) ridx = neg_idx[(size_t)t * NEGS + (ctx - 1)];
    if (ctx < NEGS) {  // warm L2 for the next gathered row (global_prefetch_b8)
      int nridx = neg_idx[(size_t)t * NEGS + ctx];
      __builtin_prefetch(out_emb + (size_t)nridx * EMB, 0, 1);
    }

    // ---- A operand: 16 context embeddings as rows (16x32 f16 per K-chunk).
    // lane m<16: row m, K = chunk*32 + 0..15
    // lane 16+m:        K = chunk*32 + 16..31
    const float* arow = out_emb + (size_t)ridx * EMB + kh * 16;
    v8f c8 = {};
    #pragma unroll
    for (int c = 0; c < 4; ++c) {
      v16h a;
      load8_cvt<0>(a, arow + c * 32 + 0);
      load8_cvt<8>(a, arow + c * 32 + 8);
      c8 = __builtin_amdgcn_wmma_f32_16x16x32_f16(
          /*neg_a=*/false, a, /*neg_b=*/false, b[c],
          /*c_mod=*/(short)0, c8, /*reuse_a=*/false, /*reuse_b=*/false);
    }

    // Diagonal of D: lanes 0..7 -> c8[lane] (M=N=lane);
    //                lanes 24..31 -> c8[lane-24] (M=N=lane-16).
    float d = c8[0];
    #pragma unroll
    for (int r = 1; r < 8; ++r) d = ((lane & 7) == r) ? c8[r] : d;

    // -log(sigmoid(+d)) = max(-d,0) + log(1+e^{-|d|})   (positive sample)
    // -log(sigmoid(-d)) = max(+d,0) + log(1+e^{-|d|})   (negative samples)
    // Shared transcendental part via v_exp_f32 / v_log_f32; branchless.
    float l = __logf(1.0f + __expf(-fabsf(d)));
    float m = fmaxf((ctx == 0) ? -d : d, 0.0f);
    acc += diag_mask * (m + l);
  }

  // ---- block reduction to one partial per block
  __shared__ float red[256];
  red[tid] = acc;
  __syncthreads();
  #pragma unroll
  for (int s = 128; s > 0; s >>= 1) {
    if (tid < s) red[tid] += red[tid + s];
    __syncthreads();
  }
  if (tid == 0) partial[blockIdx.x] = red[0];
}

__global__ void __launch_bounds__(256)
skipgram_reduce(const float* __restrict__ partial, int n, float inv_batch,
                float* __restrict__ out) {
  __shared__ float red[256];
  float s = 0.0f;
  for (int i = threadIdx.x; i < n; i += 256) s += partial[i];
  red[threadIdx.x] = s;
  __syncthreads();
  #pragma unroll
  for (int k = 128; k > 0; k >>= 1) {
    if (threadIdx.x < k) red[threadIdx.x] += red[threadIdx.x + k];
    __syncthreads();
  }
  if (threadIdx.x == 0) out[0] = red[0] * inv_batch;   // mean of -(pos+neg) losses
}

extern "C" void kernel_launch(void* const* d_in, const int* in_sizes, int n_in,
                              void* d_out, int out_size, void* d_ws, size_t ws_size,
                              hipStream_t stream) {
  const int*   in_idx  = (const int*)d_in[0];
  const int*   out_idx = (const int*)d_in[1];
  const int*   neg_idx = (const int*)d_in[2];
  const float* in_emb  = (const float*)d_in[3];
  const float* out_emb = (const float*)d_in[4];
  (void)n_in; (void)out_size; (void)ws_size;

  const int B      = in_sizes[0];            // 65536
  const int blocks = B / ROWS_PER_BLOCK;     // 512
  float* partial   = (float*)d_ws;           // 512 floats, fully overwritten each call

  skipgram_nsloss_wmma<<<blocks, 256, 0, stream>>>(in_idx, out_idx, neg_idx,
                                                   in_emb, out_emb, partial);
  skipgram_reduce<<<1, 256, 0, stream>>>(partial, blocks, 1.0f / (float)B,
                                         (float*)d_out);
}
